// KimiDeltaAttention_2714419331392
// MI455X (gfx1250) — matrix operand order repaired and checked by reading, not compiled, not run on hardware
//
#include <hip/hip_runtime.h>
#include <hip/hip_bf16.h>

// ---------------- problem constants ----------------
#define BATCH 4
#define TLEN  2048
#define HID   2048
#define NH    16
#define DK    128
#define DV    128
#define MROWS (BATCH * TLEN)          // 8192

// GEMM tiling: block 256(M) x 128(N), 8 waves, wave tile 64x64, K-slab 32
#define BM 256
#define BN 128
#define BK 32
#define LPAD 8
#define LSTR (BK + LPAD)              // 40 bf16 per LDS row (80B) -> conflict-free

typedef __attribute__((ext_vector_type(16))) __bf16 v16bf;
typedef __attribute__((ext_vector_type(8)))  __bf16 v8bf;
typedef __attribute__((ext_vector_type(8)))  float  v8f;

// ---------------- f32 -> bf16 convert ----------------
__global__ void cvt_f32_to_bf16(const float* __restrict__ in,
                                __bf16* __restrict__ out, size_t n) {
  size_t i = (size_t)blockIdx.x * blockDim.x + threadIdx.x;
  if (i < n) out[i] = (__bf16)in[i];
}

// ---- CDNA5 async copy: global -> LDS, 16B per lane, tracked by ASYNCcnt ----
__device__ __forceinline__ void async_b128_to_lds(const __bf16* gptr, void* lptr) {
  unsigned lds_off = (unsigned)(uintptr_t)lptr;            // LDS aperture: addr[31:0] = offset
  unsigned long long ga = (unsigned long long)(uintptr_t)gptr;
  asm volatile("global_load_async_to_lds_b128 %0, %1, off"
               :: "v"(lds_off), "v"(ga) : "memory");
}
__device__ __forceinline__ void wait_asynccnt0() {
  asm volatile("s_wait_asynccnt 0x0" ::: "memory");
}

// ---------------- WMMA bf16 GEMM: C[m,n] = act(sum_k A[m,k]*W[n,k] + bias[n]) ----
// A: [M,K] bf16 row-major; W: [N,K] bf16 row-major (computes A @ W^T).
// 256 threads = 8 waves (4x2), wave tile 64x64 => 16 WMMAs per K-slab from
// 8 fragment loads. K-slabs double-buffered in LDS via
// global_load_async_to_lds_b128 (ASYNCcnt), fragments via ds_load_b128.
// act: 0 = none, 1 = silu
__global__ __launch_bounds__(256)
void gemm_wmma_bf16(const __bf16* __restrict__ A, const __bf16* __restrict__ W,
                    float* __restrict__ outF, __bf16* __restrict__ outB,
                    const float* __restrict__ bias,
                    int Mdim, int Ndim, int Kdim, int act)
{
  __shared__ __bf16 sA[2][BM * LSTR];   // 2 * 256*40*2B = 40 KB
  __shared__ __bf16 sB[2][BN * LSTR];   // 2 * 128*40*2B = 20 KB

  const int bN   = blockIdx.x * BN;
  const int bM   = blockIdx.y * BM;
  const int wave = threadIdx.x >> 5;       // 0..7
  const int lane = threadIdx.x & 31;
  const int wm   = (wave & 3) * 64;        // wave M offset in block
  const int wn   = (wave >> 2) * 64;       // wave N offset in block
  const int lm   = lane & 15;
  const int lhi  = lane >> 4;              // 0/1

  v8f acc[4][4];
  const v8f vzero = {0.f,0.f,0.f,0.f,0.f,0.f,0.f,0.f};
  #pragma unroll
  for (int i = 0; i < 4; ++i)
    #pragma unroll
    for (int j = 0; j < 4; ++j) acc[i][j] = vzero;

  const int nslab = Kdim / BK;

  // per-thread slab-fill: A has 256 rows * 4 chunks = 1024 16B-chunks (4/thread),
  // B has 128 rows * 4 chunks = 512 chunks (2/thread).
  auto fill_slab = [&](int s, int buf) {
    const int k0 = s * BK;
    #pragma unroll
    for (int c = 0; c < 4; ++c) {
      int chunk = (int)threadIdx.x * 4 + c;
      int row = chunk >> 2;
      int col = (chunk & 3) * 8;
      async_b128_to_lds(A + (size_t)(bM + row) * Kdim + k0 + col,
                        &sA[buf][row * LSTR + col]);
    }
    #pragma unroll
    for (int c = 0; c < 2; ++c) {
      int chunk = (int)threadIdx.x * 2 + c;
      int row = chunk >> 2;
      int col = (chunk & 3) * 8;
      async_b128_to_lds(W + (size_t)(bN + row) * Kdim + k0 + col,
                        &sB[buf][row * LSTR + col]);
    }
  };

  fill_slab(0, 0);
  int cur = 0;

  for (int s = 0; s < nslab; ++s) {
    wait_asynccnt0();        // this wave's fills for slab s have landed
    __syncthreads();         // every wave's fills have landed
    if (s + 1 < nslab) fill_slab(s + 1, cur ^ 1);
    if (s + 2 < nslab) {     // L2 warmup two slabs ahead
      const int kp = (s + 2) * BK;
      __builtin_prefetch(A + (size_t)(bM + threadIdx.x) * Kdim + kp, 0, 1);
      __builtin_prefetch(W + (size_t)(bN + (threadIdx.x >> 1)) * Kdim + kp, 0, 1);
    }

    // ---- A fragments 16x32 (MxK), ISA lane layout:
    // lanes 0-15 : K 0..7 in elems 0..7, K 16..23 in elems 8..15
    // lanes 16-31: K 8..15 / 24..31
    v16bf a[4];
    #pragma unroll
    for (int mi = 0; mi < 4; ++mi) {
      const __bf16* p = &sA[cur][(wm + mi * 16 + lm) * LSTR + lhi * 8];
      v8bf lo = *(const v8bf*)(p);
      v8bf hi = *(const v8bf*)(p + 16);
      a[mi] = __builtin_shufflevector(lo, hi, 0,1,2,3,4,5,6,7,8,9,10,11,12,13,14,15);
    }
    // ---- B fragments 32x16 (KxN): lane = column; lanes 0-15 K 0..7 / 16..23,
    // lanes 16-31 K 8..15 / 24..31 (same order as A).
    v16bf b[4];
    #pragma unroll
    for (int ni = 0; ni < 4; ++ni) {
      const __bf16* q = &sB[cur][(wn + ni * 16 + lm) * LSTR + lhi * 8];
      v8bf blo = *(const v8bf*)(q);
      v8bf bhi = *(const v8bf*)(q + 16);
      b[ni] = __builtin_shufflevector(blo, bhi, 0,1,2,3,4,5,6,7,8,9,10,11,12,13,14,15);
    }

    #pragma unroll
    for (int mi = 0; mi < 4; ++mi)
      #pragma unroll
      for (int ni = 0; ni < 4; ++ni)
        acc[mi][ni] = __builtin_amdgcn_wmma_f32_16x16x32_bf16(
            false, a[mi], false, b[ni], (short)0, acc[mi][ni], false, false);

    __syncthreads();         // computes done before buffer reuse
    cur ^= 1;
  }

  // ---- epilogue: D layout: VGPR r -> lanes 0-15: M=r,N=lane ; lanes 16-31: M=8+r
  #pragma unroll
  for (int mi = 0; mi < 4; ++mi) {
    #pragma unroll
    for (int ni = 0; ni < 4; ++ni) {
      #pragma unroll
      for (int r = 0; r < 8; ++r) {
        int m = bM + wm + mi * 16 + r + lhi * 8;
        int n = bN + wn + ni * 16 + lm;
        float v = acc[mi][ni][r];
        if (bias) v += bias[n];
        if (act == 1) v = v / (1.f + __expf(-v));   // silu
        size_t idx = (size_t)m * Ndim + n;
        if (outB) outB[idx] = (__bf16)v;
        else      outF[idx] = v;
      }
    }
  }
}

// ---------------- beta = sigmoid(h @ b_w^T), one wave per (row, head) ------------
__global__ __launch_bounds__(256)
void beta_kernel(const float* __restrict__ h, const float* __restrict__ bw,
                 float* __restrict__ beta)
{
  int gw   = blockIdx.x * 8 + (threadIdx.x >> 5);   // global wave id
  int lane = threadIdx.x & 31;
  int r    = gw >> 4;
  int hh   = gw & 15;
  const float* hp = h  + (size_t)r  * HID;
  const float* wp = bw + (size_t)hh * HID;
  float s = 0.f;
  for (int k = lane; k < HID; k += 32) s += hp[k] * wp[k];
  #pragma unroll
  for (int off = 16; off > 0; off >>= 1) s += __shfl_xor(s, off, 32);
  if (lane == 0) beta[(size_t)r * NH + hh] = 1.f / (1.f + __expf(-s));
}

// ---------------- g = -exp(A_log[h]) * softplus(g + dt_bias[c]) ------------------
__global__ void kda_gate_kernel(float* __restrict__ g, const float* __restrict__ A_log,
                                const float* __restrict__ dt_bias, size_t n)
{
  size_t i = (size_t)blockIdx.x * blockDim.x + threadIdx.x;
  if (i >= n) return;
  int c  = (int)(i & (HID - 1));
  int hh = c >> 7;
  float x  = g[i] + dt_bias[c];
  float sp = (x > 20.f) ? x : __logf(1.f + __expf(x));
  g[i] = -__expf(A_log[hh]) * sp;
}

// ---------------- gated delta-rule scan: one block per (b,h), lane = v column ----
// State column S[:,v] lives in 128 VGPRs per lane (fully unrolled k loops);
// per-step q/k/exp(g) broadcast through LDS, read back as float4 (ds_load_b128).
__global__ __launch_bounds__(128)
void kda_scan_kernel(const __bf16* __restrict__ qbf, const __bf16* __restrict__ kbf,
                     const __bf16* __restrict__ vbf, const float* __restrict__ g,
                     const float* __restrict__ beta, const float* __restrict__ S0,
                     float* __restrict__ O, float* __restrict__ Sout)
{
  const int bh  = blockIdx.x;          // 0..63
  const int b   = bh >> 4;
  const int hh  = bh & 15;
  const int tid = threadIdx.x;         // 0..127 = v column

  __shared__ __align__(16) float qs[DK];
  __shared__ __align__(16) float ks[DK];
  __shared__ __align__(16) float egs[DK];
  __shared__ __align__(16) float red[DK];
  __shared__ float bshare;

  float S[DK];
  const float* s0 = S0 + (size_t)bh * DK * DV;
  #pragma unroll
  for (int k = 0; k < DK; ++k) S[k] = s0[(size_t)k * DV + tid];

  const float scale = 0.08838834764831845f;   // DK^-0.5

  for (int t = 0; t < TLEN; ++t) {
    size_t r    = (size_t)b * TLEN + t;
    size_t base = r * (size_t)(NH * DK) + (size_t)hh * DK;

    float qv = (float)qbf[base + tid];
    float kv = (float)kbf[base + tid];
    float vv = (float)vbf[base + tid];
    float gv = g[base + tid];

    // l2 norm of q over DK
    red[tid] = qv * qv; __syncthreads();
    for (int s = 64; s > 0; s >>= 1) { if (tid < s) red[tid] += red[tid + s]; __syncthreads(); }
    float qsum = red[0]; __syncthreads();
    // l2 norm of k over DK
    red[tid] = kv * kv; __syncthreads();
    for (int s = 64; s > 0; s >>= 1) { if (tid < s) red[tid] += red[tid + s]; __syncthreads(); }
    float ksum = red[0]; __syncthreads();

    qs[tid]  = qv * rsqrtf(qsum + 1e-6f) * scale;
    ks[tid]  = kv * rsqrtf(ksum + 1e-6f);
    egs[tid] = __expf(gv);
    if (tid == 0) bshare = beta[r * NH + hh];
    __syncthreads();

    float bt    = bshare;
    float kvacc = 0.f;
    #pragma unroll
    for (int k4 = 0; k4 < DK / 4; ++k4) {        // decay + k^T S readout
      const float4 eg = *(const float4*)&egs[k4 * 4];
      const float4 kk = *(const float4*)&ks[k4 * 4];
      float s0_ = S[4*k4+0] * eg.x; S[4*k4+0] = s0_; kvacc += kk.x * s0_;
      float s1_ = S[4*k4+1] * eg.y; S[4*k4+1] = s1_; kvacc += kk.y * s1_;
      float s2_ = S[4*k4+2] * eg.z; S[4*k4+2] = s2_; kvacc += kk.z * s2_;
      float s3_ = S[4*k4+3] * eg.w; S[4*k4+3] = s3_; kvacc += kk.w * s3_;
    }
    float delta = (vv - kvacc) * bt;
    float o = 0.f;
    #pragma unroll
    for (int k4 = 0; k4 < DK / 4; ++k4) {        // rank-1 update + (q*scale)^T S
      const float4 kk = *(const float4*)&ks[k4 * 4];
      const float4 qq = *(const float4*)&qs[k4 * 4];
      float s0_ = S[4*k4+0] + kk.x * delta; S[4*k4+0] = s0_; o += qq.x * s0_;
      float s1_ = S[4*k4+1] + kk.y * delta; S[4*k4+1] = s1_; o += qq.y * s1_;
      float s2_ = S[4*k4+2] + kk.z * delta; S[4*k4+2] = s2_; o += qq.z * s2_;
      float s3_ = S[4*k4+3] + kk.w * delta; S[4*k4+3] = s3_; o += qq.w * s3_;
    }
    O[base + tid] = o;
    __syncthreads();                     // protect shared arrays for next t
  }

  float* so = Sout + (size_t)bh * DK * DV;
  #pragma unroll
  for (int k = 0; k < DK; ++k) so[(size_t)k * DV + tid] = S[k];
}

// ---------------- o = rmsnorm(o)*w*sigmoid(factor) -> bf16, one block per (r,h) --
__global__ __launch_bounds__(128)
void out_gate_kernel(const float* __restrict__ O, const float* __restrict__ factor,
                     const float* __restrict__ onw, __bf16* __restrict__ OG)
{
  int rh  = blockIdx.x;
  int r   = rh >> 4;
  int hh  = rh & 15;
  int tid = threadIdx.x;
  size_t base = (size_t)r * (NH * DV) + (size_t)hh * DV;

  float o = O[base + tid];
  __shared__ float red[DV];
  red[tid] = o * o; __syncthreads();
  for (int s = 64; s > 0; s >>= 1) { if (tid < s) red[tid] += red[tid + s]; __syncthreads(); }
  float ms = red[0] * (1.f / (float)DV);

  float f   = factor[base + tid];
  float val = o * rsqrtf(ms + 1e-5f) * onw[tid] * (1.f / (1.f + __expf(-f)));
  OG[base + tid] = (__bf16)val;
}

// =============================== host launcher ===================================
extern "C" void kernel_launch(void* const* d_in, const int* in_sizes, int n_in,
                              void* d_out, int out_size, void* d_ws, size_t ws_size,
                              hipStream_t stream)
{
  const float* h       = (const float*)d_in[0];
  const float* S0      = (const float*)d_in[1];
  const float* q_w     = (const float*)d_in[2];
  const float* k_w     = (const float*)d_in[3];
  const float* v_w     = (const float*)d_in[4];
  const float* f_w0    = (const float*)d_in[5];
  const float* f_w1    = (const float*)d_in[6];
  const float* b_w     = (const float*)d_in[7];
  const float* A_log   = (const float*)d_in[8];
  const float* dt_bias = (const float*)d_in[9];
  const float* g_w0    = (const float*)d_in[10];
  const float* g_w1    = (const float*)d_in[11];
  const float* g_b1    = (const float*)d_in[12];
  const float* o_norm  = (const float*)d_in[13];
  const float* o_w     = (const float*)d_in[14];

  float* out  = (float*)d_out;                       // [8192, 2048]
  float* Sout = out + (size_t)MROWS * HID;           // [4,16,128,128]

  // ---------------- workspace layout ----------------
  char* ws = (char*)d_ws;
  size_t off = 0;
  auto take = [&](size_t bytes) { char* p = ws + off; off += bytes; return p; };

  __bf16* HBF  = (__bf16*)take((size_t)MROWS * HID * 2);     // h bf16; reused as OGBF
  __bf16* WQ   = (__bf16*)take((size_t)HID * HID * 2);
  __bf16* WK   = (__bf16*)take((size_t)HID * HID * 2);
  __bf16* WV   = (__bf16*)take((size_t)HID * HID * 2);
  __bf16* WO   = (__bf16*)take((size_t)HID * HID * 2);
  __bf16* WF0  = (__bf16*)take((size_t)DV * HID * 2);
  __bf16* WF1  = (__bf16*)take((size_t)(NH * DK) * DV * 2);
  __bf16* WG0  = (__bf16*)take((size_t)DV * HID * 2);
  __bf16* WG1  = (__bf16*)take((size_t)(NH * DV) * DV * 2);
  __bf16* QBF  = (__bf16*)take((size_t)MROWS * HID * 2);
  __bf16* KBF  = (__bf16*)take((size_t)MROWS * HID * 2);
  __bf16* VBF  = (__bf16*)take((size_t)MROWS * HID * 2);
  float*  GF   = (float*)take((size_t)MROWS * HID * 4);      // g, then factor
  float*  BETA = (float*)take((size_t)MROWS * NH * 4);
  __bf16* T128 = (__bf16*)take((size_t)MROWS * DV * 2);
  float*  OF   = (float*)take((size_t)MROWS * HID * 4);
  __bf16* OGBF = HBF;                                        // reuse after h consumed

  auto cvt = [&](const float* src, __bf16* dst, size_t n) {
    cvt_f32_to_bf16<<<dim3((unsigned)((n + 255) / 256)), dim3(256), 0, stream>>>(src, dst, n);
  };

  // 1) bf16 conversions
  cvt(h,    HBF, (size_t)MROWS * HID);
  cvt(q_w,  WQ,  (size_t)HID * HID);
  cvt(k_w,  WK,  (size_t)HID * HID);
  cvt(v_w,  WV,  (size_t)HID * HID);
  cvt(o_w,  WO,  (size_t)HID * HID);
  cvt(f_w0, WF0, (size_t)DV * HID);
  cvt(f_w1, WF1, (size_t)(NH * DK) * DV);
  cvt(g_w0, WG0, (size_t)DV * HID);
  cvt(g_w1, WG1, (size_t)(NH * DV) * DV);

  dim3 blk(256);
  dim3 gFull(HID / BN, MROWS / BM);     // (16, 32)
  dim3 gNarrow(1, MROWS / BM);          // N = 128

  // 2) q/k/v projections with fused silu -> bf16
  gemm_wmma_bf16<<<gFull, blk, 0, stream>>>(HBF, WQ, nullptr, QBF, nullptr, MROWS, HID, HID, 1);
  gemm_wmma_bf16<<<gFull, blk, 0, stream>>>(HBF, WK, nullptr, KBF, nullptr, MROWS, HID, HID, 1);
  gemm_wmma_bf16<<<gFull, blk, 0, stream>>>(HBF, WV, nullptr, VBF, nullptr, MROWS, HID, HID, 1);

  // 3) gate path: g = (h @ f_w0^T) @ f_w1^T, then KDA gate transform
  gemm_wmma_bf16<<<gNarrow, blk, 0, stream>>>(HBF, WF0, nullptr, T128, nullptr, MROWS, DV, HID, 0);
  gemm_wmma_bf16<<<gFull,   blk, 0, stream>>>(T128, WF1, GF, nullptr, nullptr, MROWS, HID, DV, 0);
  {
    size_t n = (size_t)MROWS * HID;
    kda_gate_kernel<<<dim3((unsigned)((n + 255) / 256)), dim3(256), 0, stream>>>(GF, A_log, dt_bias, n);
  }

  // 4) beta
  beta_kernel<<<dim3(MROWS * NH / 8), blk, 0, stream>>>(h, b_w, BETA);

  // 5) sequential gated delta-rule scan (state in VGPRs), writes O and final state
  kda_scan_kernel<<<dim3(BATCH * NH), dim3(128), 0, stream>>>(QBF, KBF, VBF, GF, BETA, S0, OF, Sout);

  // 6) output-gate path: factor = (h @ g_w0^T) @ g_w1^T + g_b1 (reuses GF after scan)
  gemm_wmma_bf16<<<gNarrow, blk, 0, stream>>>(HBF, WG0, nullptr, T128, nullptr, MROWS, DV, HID, 0);
  gemm_wmma_bf16<<<gFull,   blk, 0, stream>>>(T128, WG1, GF, nullptr, g_b1, MROWS, HID, DV, 0);

  // 7) rmsnorm + sigmoid gate -> bf16 (reuses HBF storage; h fully consumed)
  out_gate_kernel<<<dim3(MROWS * NH), dim3(128), 0, stream>>>(OF, GF, o_norm, OGBF);

  // 8) final projection: out = o_gated @ o_w^T (f32 out)
  gemm_wmma_bf16<<<gFull, blk, 0, stream>>>(OGBF, WO, out, nullptr, nullptr, MROWS, HID, HID, 0);
}
